// TDMPC_30056181137686
// MI455X (gfx1250) — matrix operand order repaired and checked by reading, not compiled, not run on hardware
//
#include <hip/hip_runtime.h>
#include <math.h>

typedef _Float16 h8  __attribute__((ext_vector_type(8)));
typedef _Float16 h16 __attribute__((ext_vector_type(16)));
typedef float    f8  __attribute__((ext_vector_type(8)));

#define NSAMP   65536
#define HOR     5
#define LAT     50
#define ADIM    4
#define HID     256
#define MTILE   32
#define NWAVES  8
#define NTHR    256
#define NELITE  64

__device__ __forceinline__ h16 cat16(h8 lo, h8 hi) {
  return __builtin_shufflevector(lo, hi, 0,1,2,3,4,5,6,7,8,9,10,11,12,13,14,15);
}

// ---------------------------------------------------------------------------
// Y[MTILE x Npad] = act( X[MTILE x Kpad] @ Wpacked + bias ), f16 in LDS,
// f32 WMMA accumulate. Each wave owns one N-tile and computes BOTH M-tiles
// with a single B-fragment load (2 independent accumulator chains).
// Wpacked layout: [n_tile][k_tile][lane(32)][16 halves] = CDNA5 B-fragment.
// ---------------------------------------------------------------------------
__device__ __forceinline__ void gemm_layer(
    const _Float16* __restrict__ X, int xs,
    const _Float16* __restrict__ Wp, const float* __restrict__ bias,
    int Kpad, int Npad,
    _Float16* __restrict__ Y, int ys,
    bool relu, int wave, int lane)
{
  const int ntiles = Npad >> 4;
  const int ktiles = Kpad >> 5;
  const int lgrp = lane >> 4;   // 0 or 1
  const int lmod = lane & 15;
  const _Float16* x0 = X + lmod * xs + lgrp * 8;          // rows 0..15
  const _Float16* x1 = X + (16 + lmod) * xs + lgrp * 8;   // rows 16..31
  for (int nt = wave; nt < ntiles; nt += NWAVES) {
    const _Float16* wl = Wp + ((size_t)(nt * ktiles) << 9) + ((size_t)lane << 4);
    f8 c0 = {0.f,0.f,0.f,0.f,0.f,0.f,0.f,0.f};
    f8 c1 = {0.f,0.f,0.f,0.f,0.f,0.f,0.f,0.f};
    for (int kt = 0; kt < ktiles; ++kt) {
      h8 blo = *(const h8*)(wl + (kt << 9));
      h8 bhi = *(const h8*)(wl + (kt << 9) + 8);
      h8 a0l = *(const h8*)(x0 + (kt << 5));        // K = 32kt+8g .. +7
      h8 a0h = *(const h8*)(x0 + (kt << 5) + 16);   // K = 32kt+16+8g .. +7
      h8 a1l = *(const h8*)(x1 + (kt << 5));
      h8 a1h = *(const h8*)(x1 + (kt << 5) + 16);
      h16 b  = cat16(blo, bhi);
      h16 a0 = cat16(a0l, a0h);
      h16 a1 = cat16(a1l, a1h);
      c0 = __builtin_amdgcn_wmma_f32_16x16x32_f16(false, a0, false, b,
                                                  (short)0, c0, false, false);
      c1 = __builtin_amdgcn_wmma_f32_16x16x32_f16(false, a1, false, b,
                                                  (short)0, c1, false, false);
    }
    const float bv = bias[nt*16 + lmod];
    _Float16* y0 = Y + (lgrp << 3) * ys + nt*16 + lmod;
    _Float16* y1 = Y + (16 + (lgrp << 3)) * ys + nt*16 + lmod;
    #pragma unroll
    for (int r = 0; r < 8; ++r) {       // row m = 8*lgrp + r  (C/D layout)
      float v0 = c0[r] + bv;
      float v1 = c1[r] + bv;
      if (relu) { v0 = fmaxf(v0, 0.f); v1 = fmaxf(v1, 0.f); }
      y0[r * ys] = (_Float16)v0;
      y1[r * ys] = (_Float16)v1;
    }
  }
}

// ---------------------------------------------------------------------------
// Weight packing: f32 row-major (Ksrc x Nsrc) -> f16 B-fragment stream,
// padded with zeros to (32*ktiles x Npad).
// ---------------------------------------------------------------------------
__global__ __launch_bounds__(256) void pack_kernel(
    const float* __restrict__ src, int Ksrc, int Nsrc, int ktiles,
    _Float16* __restrict__ dst, int total)
{
  int idx = blockIdx.x * 256 + threadIdx.x;
  if (idx >= total) return;
  int frag = idx >> 9;
  int lane = (idx >> 4) & 31;
  int j    = idx & 15;
  int nt = frag / ktiles;
  int kt = frag - nt * ktiles;
  int n = nt * 16 + (lane & 15);
  int k = kt * 32 + ((lane >> 4) << 3) + (j & 7) + ((j >> 3) << 4);
  float v = (k < Ksrc && n < Nsrc) ? src[(size_t)k * Nsrc + n] : 0.f;
  dst[idx] = (_Float16)v;
}

__global__ void pad_bias_kernel(const float* __restrict__ src, int N,
                                float* __restrict__ dst)
{
  int i = threadIdx.x;
  dst[i] = (i < N) ? src[i] : 0.f;
}

// ---------------------------------------------------------------------------
// Encoder: single observation through 39->256->256->50 (scalar f32, 1 block)
// ---------------------------------------------------------------------------
__global__ __launch_bounds__(256) void enc_kernel(
    const float* __restrict__ obs,
    const float* __restrict__ W1, const float* __restrict__ b1,
    const float* __restrict__ W2, const float* __restrict__ b2,
    const float* __restrict__ W3, const float* __restrict__ b3,
    float* __restrict__ z0)
{
  __shared__ float o[40];
  __shared__ float h1[HID];
  __shared__ float h2[HID];
  int tid = threadIdx.x;
  if (tid < 39) o[tid] = obs[tid];
  __syncthreads();
  float h = b1[tid];
  for (int i = 0; i < 39; ++i) h += o[i] * W1[i*HID + tid];
  h1[tid] = fmaxf(h, 0.f);
  __syncthreads();
  h = b2[tid];
  for (int i = 0; i < HID; ++i) h += h1[i] * W2[i*HID + tid];
  h2[tid] = fmaxf(h, 0.f);
  __syncthreads();
  if (tid < LAT) {
    float z = b3[tid];
    for (int i = 0; i < HID; ++i) z += h2[i] * W3[i*LAT + tid];
    z0[tid] = z;
  }
}

// ---------------------------------------------------------------------------
// Main rollout: each block handles 32 samples through the 5-step rollout.
// ---------------------------------------------------------------------------
struct RP {
  const float* actions;
  const float* z0;
  const _Float16 *dW1,*dW2,*dW3, *rW1,*rW2,*rW3, *pW1,*pW2,*pW3;
  const _Float16 *aW1,*aW2,*aW3, *bW1,*bW2,*bW3;           // q1, q2
  const float *db1,*db2,*db3, *rb1,*rb2,*rb3, *pb1,*pb2,*pb3;
  const float *ab1,*ab2,*ab3, *bb1,*bb2,*bb3;
  float* values;
};

__global__ __launch_bounds__(NTHR) void rollout_kernel(RP p)
{
  __shared__ __align__(16) _Float16 Xb[MTILE*64];
  __shared__ __align__(16) _Float16 H1[MTILE*HID];
  __shared__ __align__(16) _Float16 H2[MTILE*HID];
  __shared__ __align__(16) _Float16 T3[MTILE*16];
  __shared__ float G[MTILE];
  __shared__ float Q1[MTILE];
  const int tid  = threadIdx.x;
  const int lane = tid & 31;
  const int wave = tid >> 5;
  const int base = blockIdx.x * MTILE;

  // X = [z0 | a_0 | zeros]
  for (int i = tid; i < MTILE*64; i += NTHR) {
    int m = i >> 6, c = i & 63;
    float v;
    if (c < LAT)            v = p.z0[c];
    else if (c < LAT+ADIM)  v = p.actions[(size_t)(base+m)*ADIM + (c-LAT)];
    else                    v = 0.f;
    Xb[i] = (_Float16)v;
  }
  if (tid < MTILE) G[tid] = 0.f;
  __syncthreads();

  float disc = 1.f;
  for (int t = 0; t < HOR; ++t) {
    // reward head: G += disc * rew(x)
    gemm_layer(Xb,64,  p.rW1,p.rb1, 64,HID,  H1,HID, true,  wave,lane); __syncthreads();
    gemm_layer(H1,HID, p.rW2,p.rb2, HID,HID, H2,HID, true,  wave,lane); __syncthreads();
    gemm_layer(H2,HID, p.rW3,p.rb3, HID,16,  T3,16,  false, wave,lane); __syncthreads();
    if (tid < MTILE) G[tid] += disc * (float)T3[tid*16];
    // dynamics: z' = dyn(x)  (writes z' + zero padding back into X)
    gemm_layer(Xb,64,  p.dW1,p.db1, 64,HID,  H1,HID, true,  wave,lane); __syncthreads();
    gemm_layer(H1,HID, p.dW2,p.db2, HID,HID, H2,HID, true,  wave,lane); __syncthreads();
    gemm_layer(H2,HID, p.dW3,p.db3, HID,64,  Xb,64,  false, wave,lane); __syncthreads();
    if (t < HOR-1) {
      if (tid < MTILE*ADIM) {
        int m = tid >> 2, a = tid & 3;
        Xb[m*64 + LAT + a] =
          (_Float16)p.actions[((size_t)(t+1)*NSAMP + base + m)*ADIM + a];
      }
      __syncthreads();
    }
    disc *= 0.99f;
  }
  // policy: a_term = tanh(pi(z))   (X holds [z | zeros])
  gemm_layer(Xb,64,  p.pW1,p.pb1, 64,HID,  H1,HID, true,  wave,lane); __syncthreads();
  gemm_layer(H1,HID, p.pW2,p.pb2, HID,HID, H2,HID, true,  wave,lane); __syncthreads();
  gemm_layer(H2,HID, p.pW3,p.pb3, HID,16,  T3,16,  false, wave,lane); __syncthreads();
  if (tid < MTILE*ADIM) {
    int m = tid >> 2, a = tid & 3;
    Xb[m*64 + LAT + a] = (_Float16)tanhf((float)T3[m*16 + a]);
  }
  __syncthreads();
  // q1
  gemm_layer(Xb,64,  p.aW1,p.ab1, 64,HID,  H1,HID, true,  wave,lane); __syncthreads();
  gemm_layer(H1,HID, p.aW2,p.ab2, HID,HID, H2,HID, true,  wave,lane); __syncthreads();
  gemm_layer(H2,HID, p.aW3,p.ab3, HID,16,  T3,16,  false, wave,lane); __syncthreads();
  if (tid < MTILE) Q1[tid] = (float)T3[tid*16];
  __syncthreads();
  // q2
  gemm_layer(Xb,64,  p.bW1,p.bb1, 64,HID,  H1,HID, true,  wave,lane); __syncthreads();
  gemm_layer(H1,HID, p.bW2,p.bb2, HID,HID, H2,HID, true,  wave,lane); __syncthreads();
  gemm_layer(H2,HID, p.bW3,p.bb3, HID,16,  T3,16,  false, wave,lane); __syncthreads();
  if (tid < MTILE) {
    float q2v = (float)T3[tid*16];
    p.values[base + tid] = G[tid] + disc * fminf(Q1[tid], q2v);  // disc = 0.99^5
  }
}

// ---------------------------------------------------------------------------
// Exact top-64 (tie -> lowest index), softmax(T=0.5), weighted action mean.
// Destroys `values` (it is recomputed every call).
// ---------------------------------------------------------------------------
__global__ __launch_bounds__(1024) void topk_kernel(
    float* __restrict__ values, const float* __restrict__ actions,
    float* __restrict__ out)
{
  __shared__ float sv[1024];
  __shared__ int   si[1024];
  __shared__ float ev[NELITE];
  __shared__ int   ei[NELITE];
  __shared__ float ssum;
  const int tid = threadIdx.x;
  for (int it = 0; it < NELITE; ++it) {
    float bv = -3.0e38f; int bi = 0x7fffffff;
    for (int j = tid; j < NSAMP; j += 1024) {
      float v = values[j];
      if (v > bv || (v == bv && j < bi)) { bv = v; bi = j; }
    }
    sv[tid] = bv; si[tid] = bi;
    __syncthreads();
    for (int s = 512; s > 0; s >>= 1) {
      if (tid < s) {
        float v2 = sv[tid+s]; int i2 = si[tid+s];
        if (v2 > sv[tid] || (v2 == sv[tid] && i2 < si[tid])) { sv[tid]=v2; si[tid]=i2; }
      }
      __syncthreads();
    }
    if (tid == 0) { ev[it] = sv[0]; ei[it] = si[0]; values[si[0]] = -3.0e38f; }
    __syncthreads();
  }
  if (tid < NELITE) sv[tid] = expf(0.5f * (ev[tid] - ev[0]));  // ev[0] = max
  __syncthreads();
  if (tid == 0) { float s = 0.f; for (int k = 0; k < NELITE; ++k) s += sv[k]; ssum = s; }
  __syncthreads();
  if (tid < HOR*ADIM) {
    int hh = tid >> 2, a = tid & 3;
    float acc = 0.f;
    for (int k = 0; k < NELITE; ++k)
      acc += sv[k] * actions[((size_t)hh*NSAMP + ei[k])*ADIM + a];
    out[tid] = acc / ssum;
  }
}

// ---------------------------------------------------------------------------
extern "C" void kernel_launch(void* const* d_in, const int* in_sizes, int n_in,
                              void* d_out, int out_size, void* d_ws, size_t ws_size,
                              hipStream_t stream)
{
  (void)in_sizes; (void)n_in; (void)out_size; (void)ws_size;
  const float* obs     = (const float*)d_in[0];
  const float* actions = (const float*)d_in[1];
  // enc: 2..7   dyn: 8..13   rew: 14..19   pi: 20..25   q1: 26..31   q2: 32..37
  const float** I = (const float**)d_in;

  char* w = (char*)d_ws;
  size_t off = 0;
  auto alloc = [&](size_t bytes) -> void* {
    void* p = w + off; off += (bytes + 255) & ~(size_t)255; return p;
  };
  float* z0     = (float*)alloc(64 * 4);
  float* values = (float*)alloc((size_t)NSAMP * 4);
  float* db3 = (float*)alloc(64 * 4);
  float* rb3 = (float*)alloc(16 * 4);
  float* pb3 = (float*)alloc(16 * 4);
  float* ab3 = (float*)alloc(16 * 4);
  float* bb3 = (float*)alloc(16 * 4);
  const size_t W1H = 16384;   // 64x256 packed halves
  const size_t W2H = 65536;   // 256x256
  const size_t W3D = 16384;   // 256x64  (dyn)
  const size_t W3S = 4096;    // 256x16  (rew/pi/q)
  _Float16* dW1 = (_Float16*)alloc(W1H*2);
  _Float16* dW2 = (_Float16*)alloc(W2H*2);
  _Float16* dW3 = (_Float16*)alloc(W3D*2);
  _Float16* rW1 = (_Float16*)alloc(W1H*2);
  _Float16* rW2 = (_Float16*)alloc(W2H*2);
  _Float16* rW3 = (_Float16*)alloc(W3S*2);
  _Float16* pW1 = (_Float16*)alloc(W1H*2);
  _Float16* pW2 = (_Float16*)alloc(W2H*2);
  _Float16* pW3 = (_Float16*)alloc(W3S*2);
  _Float16* aW1 = (_Float16*)alloc(W1H*2);
  _Float16* aW2 = (_Float16*)alloc(W2H*2);
  _Float16* aW3 = (_Float16*)alloc(W3S*2);
  _Float16* bW1 = (_Float16*)alloc(W1H*2);
  _Float16* bW2 = (_Float16*)alloc(W2H*2);
  _Float16* bW3 = (_Float16*)alloc(W3S*2);

  auto pack = [&](const float* src, int Ks, int Ns, int Kpad, int Npad, _Float16* dst) {
    int ktiles = Kpad / 32;
    int total  = (Npad / 16) * ktiles * 512;
    pack_kernel<<<(total + 255) / 256, 256, 0, stream>>>(src, Ks, Ns, ktiles, dst, total);
  };
  auto padb = [&](const float* src, int N, int Npad, float* dst) {
    pad_bias_kernel<<<1, Npad, 0, stream>>>(src, N, dst);
  };

  // dyn (54->256->256->50)
  pack(I[8],  54, 256, 64,  256, dW1);
  pack(I[10], 256,256, 256, 256, dW2);
  pack(I[12], 256, 50, 256, 64,  dW3);  padb(I[13], 50, 64, db3);
  // rew (54->256->256->1)
  pack(I[14], 54, 256, 64,  256, rW1);
  pack(I[16], 256,256, 256, 256, rW2);
  pack(I[18], 256, 1,  256, 16,  rW3);  padb(I[19], 1, 16, rb3);
  // pi (50->256->256->4)
  pack(I[20], 50, 256, 64,  256, pW1);
  pack(I[22], 256,256, 256, 256, pW2);
  pack(I[24], 256, 4,  256, 16,  pW3);  padb(I[25], 4, 16, pb3);
  // q1 (54->256->256->1)
  pack(I[26], 54, 256, 64,  256, aW1);
  pack(I[28], 256,256, 256, 256, aW2);
  pack(I[30], 256, 1,  256, 16,  aW3);  padb(I[31], 1, 16, ab3);
  // q2
  pack(I[32], 54, 256, 64,  256, bW1);
  pack(I[34], 256,256, 256, 256, bW2);
  pack(I[36], 256, 1,  256, 16,  bW3);  padb(I[37], 1, 16, bb3);

  // encoder -> z0
  enc_kernel<<<1, 256, 0, stream>>>(obs, I[2], I[3], I[4], I[5], I[6], I[7], z0);

  RP p;
  p.actions = actions; p.z0 = z0; p.values = values;
  p.dW1=dW1; p.dW2=dW2; p.dW3=dW3; p.db1=I[9];  p.db2=I[11]; p.db3=db3;
  p.rW1=rW1; p.rW2=rW2; p.rW3=rW3; p.rb1=I[15]; p.rb2=I[17]; p.rb3=rb3;
  p.pW1=pW1; p.pW2=pW2; p.pW3=pW3; p.pb1=I[21]; p.pb2=I[23]; p.pb3=pb3;
  p.aW1=aW1; p.aW2=aW2; p.aW3=aW3; p.ab1=I[27]; p.ab2=I[29]; p.ab3=ab3;
  p.bW1=bW1; p.bW2=bW2; p.bW3=bW3; p.bb1=I[33]; p.bb2=I[35]; p.bb3=bb3;

  rollout_kernel<<<NSAMP / MTILE, NTHR, 0, stream>>>(p);
  topk_kernel<<<1, 1024, 0, stream>>>(values, actions, (float*)d_out);
}
